// YOLOv1_84413287235957
// MI455X (gfx1250) — compile-verified, matrix operand order checked by reading, not run on hardware
//
#include <hip/hip_runtime.h>
#include <math.h>

// ---- problem constants (match reference) ----
static constexpr int P      = 9216;   // 96*96 spatial positions
static constexpr int HW     = 96;
static constexpr int CCH    = 512;    // channels (K)
static constexpr int NCLS   = 20;
static constexpr int NS     = 16384;  // padded sort length (pow2 >= P)
static constexpr int NCHUNK = CCH / 32;          // 16 K-chunks
static constexpr int FRAG_HALFS = 3 * NCHUNK * 32 * 16;   // 24576 halfs = 48KB
static constexpr float CONF    = 0.01f;
static constexpr float NMS_T   = 0.5f;
static constexpr float STRIDEF = 32.0f;

typedef __attribute__((ext_vector_type(16))) _Float16 v16h;
typedef __attribute__((ext_vector_type(8)))  float    v8f;
typedef unsigned int u32x4 __attribute__((ext_vector_type(4)));
typedef int          i32x4 __attribute__((ext_vector_type(4)));
typedef int          i32x8 __attribute__((ext_vector_type(8)));

// ---------------------------------------------------------------------------
// Pre-swizzle weights into WMMA B-fragment order, f16:
//   F[f][kc][lane][h],  f=0: W1 cols 0..15, f=1: W1 cols 16..31, f=2: W2 cols 0..15
// B fragment (32x16 f16): lane L -> column N=L%16; half h -> K = h + (h>=8?8:0) + (L>=16?8:0)
// ---------------------------------------------------------------------------
__global__ void pack_frags(const float* __restrict__ obj_w,
                           const float* __restrict__ cls_w,
                           const float* __restrict__ reg_w,
                           _Float16* __restrict__ F)
{
    int t = blockIdx.x * blockDim.x + threadIdx.x;
    if (t >= FRAG_HALFS) return;
    int h    = t & 15;
    int lane = (t >> 4) & 31;
    int kc   = (t >> 9) & 15;
    int f    = t >> 13;
    int m    = lane & 15;
    int hi   = lane >> 4;
    int K    = kc * 32 + h + ((h & 8) ? 8 : 0) + (hi ? 8 : 0);

    float v = 0.0f;
    if (f == 0) {                    // W1 cols 0..15: col0=obj, 1..15=cls[0..14]
        if (m == 0)      v = obj_w[K];
        else             v = cls_w[(m - 1) * CCH + K];
    } else if (f == 1) {             // W1 cols 16..31: 16..20 = cls[15..19]
        int n = 16 + m;
        if (n <= 20)     v = cls_w[(n - 1) * CCH + K];
    } else {                         // W2 cols 0..15: 0..3 = reg rows
        if (m < 4)       v = reg_w[m * CCH + K];
    }
    F[t] = (_Float16)v;
}

// ---------------------------------------------------------------------------
// WMMA head GEMM.  One wave (wave32) per 16-position M-tile, 8 waves/block.
// B fragments staged block-wide into LDS via the Tensor Data Mover, then
// read with ds_load_b128.  A fragments: coalesced f32 loads + cvt to f16.
// ---------------------------------------------------------------------------
__global__ void __launch_bounds__(256)
head_gemm(const float* __restrict__ clsf, const float* __restrict__ regf,
          const _Float16* __restrict__ Ffrag,
          float* __restrict__ out1, float* __restrict__ out2)
{
    __shared__ __align__(32) _Float16 shF[FRAG_HALFS];   // 48 KB

    const int lane = threadIdx.x & 31;
    const int wave = threadIdx.x >> 5;

    // ---- TDM: DMA the 48KB fragment table (12288 dwords) into LDS ----
    if (wave == 0) {
        const unsigned long long ga = (unsigned long long)(const void*)Ffrag;
        const unsigned int lds_off  = (unsigned int)(size_t)(void*)shF;
        const unsigned int ndw      = FRAG_HALFS / 2;     // 12288 dwords

        u32x4 g0;
        g0[0] = 1u;                                  // count=1, user descriptor
        g0[1] = lds_off;                             // lds_addr (bytes)
        g0[2] = (unsigned int)(ga & 0xffffffffull);  // global_addr[31:0]
        g0[3] = (unsigned int)((ga >> 32) & 0x1ffffffull) | (2u << 30); // [56:32] | type=2

        i32x8 g1;
        g1[0] = (int)(2u << 16);                     // wg_mask=0, data_size=4B
        g1[1] = (int)((ndw & 0xffffu) << 16);        // tensor_dim0[15:0]
        g1[2] = (int)((ndw >> 16) | (1u << 16));     // tensor_dim0[31:16] | tensor_dim1=1
        g1[3] = (int)((ndw & 0xffffu) << 16);        // tile_dim0 = ndw
        g1[4] = 1;                                   // tile_dim1 = 1, tile_dim2 = 0
        g1[5] = (int)ndw;                            // tensor_dim0_stride[31:0]
        g1[6] = 0;                                   // stride hi / dim1_stride lo
        g1[7] = 0;

        i32x4 g2 = {};                               // 2D tensor: groups 2/3 unused
        i32x4 g3 = {};
#if __clang_major__ >= 23
        i32x8 g4 = {};
        __builtin_amdgcn_tensor_load_to_lds(g0, g1, g2, g3, g4, 0);
#else
        __builtin_amdgcn_tensor_load_to_lds(g0, g1, g2, g3, 0);
#endif
        __builtin_amdgcn_s_wait_tensorcnt(0);
    }
    __syncthreads();

    const int tile = blockIdx.x * 8 + wave;       // 576 tiles total
    const int p0   = tile * 16;
    const int m    = lane & 15;
    const int hi   = lane >> 4;                   // 0 or 1 (half-wave)
    const int p    = p0 + m;                      // this lane's row/position

    v8f acc1a = {};   // obj+cls cols 0..15
    v8f acc1b = {};   // obj+cls cols 16..31
    v8f acc2  = {};   // reg   cols 0..15

    for (int kc = 0; kc < NCHUNK; ++kc) {
        const int k0 = kc * 32;
        if (k0 + 32 < CCH) {
            __builtin_prefetch(&clsf[(size_t)(k0 + 32) * P + p], 0, 1);
            __builtin_prefetch(&regf[(size_t)(k0 + 32) * P + p], 0, 1);
        }
        // A fragments: 16 channel loads each (coalesced across lanes), cvt f32->f16
        v16h aC, aR;
        #pragma unroll
        for (int h = 0; h < 16; ++h) {
            const int K = k0 + h + ((h & 8) ? 8 : 0) + (hi ? 8 : 0);
            aC[h] = (_Float16)clsf[(size_t)K * P + p];
            aR[h] = (_Float16)regf[(size_t)K * P + p];
        }
        // B fragments from LDS (2x ds_load_b128 each)
        const v16h b1a = *(const v16h*)&shF[((0 * NCHUNK + kc) * 32 + lane) * 16];
        const v16h b1b = *(const v16h*)&shF[((1 * NCHUNK + kc) * 32 + lane) * 16];
        const v16h b2  = *(const v16h*)&shF[((2 * NCHUNK + kc) * 32 + lane) * 16];

        acc1a = __builtin_amdgcn_wmma_f32_16x16x32_f16(false, aC, false, b1a, (short)0, acc1a, false, false);
        acc1b = __builtin_amdgcn_wmma_f32_16x16x32_f16(false, aC, false, b1b, (short)0, acc1b, false, false);
        acc2  = __builtin_amdgcn_wmma_f32_16x16x32_f16(false, aR, false, b2,  (short)0, acc2,  false, false);
    }

    #pragma unroll
    for (int r = 0; r < 8; ++r) {
        const int row = p0 + r + hi * 8;
        out1[row * 32 + m]      = acc1a[r];
        out1[row * 32 + 16 + m] = acc1b[r];
        out2[row * 16 + m]      = acc2[r];
    }
}

// ---------------------------------------------------------------------------
// Decode: sigmoid scoring, argmax label, box decode; emit sort keys.
// d_out layout: bbox[P][4] | score[P] | label[P] | keep[P]   (all float)
// ---------------------------------------------------------------------------
__device__ __forceinline__ float sigm(float x) { return 1.0f / (1.0f + expf(-x)); }

__global__ void decode_score(const float* __restrict__ out1, const float* __restrict__ out2,
                             const float* __restrict__ obj_b, const float* __restrict__ cls_b,
                             const float* __restrict__ reg_b,
                             float* __restrict__ dout,
                             float* __restrict__ key, int* __restrict__ sidx,
                             int* __restrict__ labi)
{
    int pIdx = blockIdx.x * blockDim.x + threadIdx.x;
    if (pIdx >= P) return;

    float so = sigm(out1[pIdx * 32 + 0] + obj_b[0]);
    float best = -1.0f; int bj = 0;
    #pragma unroll
    for (int j = 0; j < NCLS; ++j) {
        float s = so * sigm(out1[pIdx * 32 + 1 + j] + cls_b[j]);
        if (s > best) { best = s; bj = j; }
    }

    float r0 = out2[pIdx * 16 + 0] + reg_b[0];
    float r1 = out2[pIdx * 16 + 1] + reg_b[1];
    float r2 = out2[pIdx * 16 + 2] + reg_b[2];
    float r3 = out2[pIdx * 16 + 3] + reg_b[3];
    float gx = (float)(pIdx % HW);
    float gy = (float)(pIdx / HW);
    float cx = (sigm(r0) + gx) * STRIDEF;
    float cy = (sigm(r1) + gy) * STRIDEF;
    float w  = expf(r2) * STRIDEF;
    float h  = expf(r3) * STRIDEF;

    dout[pIdx * 4 + 0] = cx - 0.5f * w;
    dout[pIdx * 4 + 1] = cy - 0.5f * h;
    dout[pIdx * 4 + 2] = cx + 0.5f * w;
    dout[pIdx * 4 + 3] = cy + 0.5f * h;
    dout[4 * P + pIdx] = best;
    dout[5 * P + pIdx] = (float)bj;
    labi[pIdx] = bj;

    bool valid = best >= CONF;
    key[pIdx]  = valid ? best : -INFINITY;
    sidx[pIdx] = pIdx;
}

__global__ void pad_init(float* __restrict__ key, int* __restrict__ sidx)
{
    int i = P + blockIdx.x * blockDim.x + threadIdx.x;
    if (i >= NS) return;
    key[i]  = -INFINITY;
    sidx[i] = P;            // sentinel: >= P means padding
}

// ---------------------------------------------------------------------------
// Global-memory bitonic sort pass (descending by key)
// ---------------------------------------------------------------------------
__global__ void bitonic_pass(float* __restrict__ key, int* __restrict__ sidx, int k, int j)
{
    int i = blockIdx.x * blockDim.x + threadIdx.x;
    if (i >= NS) return;
    int ixj = i ^ j;
    if (ixj <= i || ixj >= NS) return;
    bool desc = ((i & k) == 0);
    float ki = key[i], kj = key[ixj];
    bool doSwap = desc ? (ki < kj) : (ki > kj);
    if (doSwap) {
        key[i] = kj; key[ixj] = ki;
        int t = sidx[i]; sidx[i] = sidx[ixj]; sidx[ixj] = t;
    }
}

// ---------------------------------------------------------------------------
// Gather sorted boxes/labels/valid into SoA for the NMS scan
// ---------------------------------------------------------------------------
__global__ void gather_sorted(const float* __restrict__ key, const int* __restrict__ sidx,
                              const float* __restrict__ dout, const int* __restrict__ labi,
                              float* __restrict__ sx1, float* __restrict__ sy1,
                              float* __restrict__ sx2, float* __restrict__ sy2,
                              int* __restrict__ slab, int* __restrict__ sval)
{
    int i = blockIdx.x * blockDim.x + threadIdx.x;
    if (i >= NS) return;
    int oi = sidx[i];
    if (oi < P) {
        sx1[i] = dout[oi * 4 + 0];
        sy1[i] = dout[oi * 4 + 1];
        sx2[i] = dout[oi * 4 + 2];
        sy2[i] = dout[oi * 4 + 3];
        slab[i] = labi[oi];
        sval[i] = (key[i] >= CONF) ? 1 : 0;
    } else {
        sx1[i] = sy1[i] = sx2[i] = sy2[i] = 0.0f;
        slab[i] = -1;
        sval[i] = 0;
    }
}

// ---------------------------------------------------------------------------
// Greedy NMS over the sorted list.  Single workgroup; keep flags in LDS;
// inner IoU checks parallel over threads.  Sorted desc => once invalid, done.
// ---------------------------------------------------------------------------
__global__ void __launch_bounds__(512)
nms_greedy(const float* __restrict__ sx1, const float* __restrict__ sy1,
           const float* __restrict__ sx2, const float* __restrict__ sy2,
           const int* __restrict__ slab, const int* __restrict__ sval,
           int* __restrict__ skeep_g)
{
    __shared__ unsigned char skeep[NS];
    __shared__ float bx1, by1, bx2, by2;
    __shared__ int blab, bval, s_flag, s_done;
    const int tid = threadIdx.x;
    const int nt  = blockDim.x;

    if (tid == 0) s_done = 0;
    for (int i = tid; i < NS; i += nt) skeep[i] = 0;
    __syncthreads();

    for (int i = 0; i < NS; ++i) {
        if (tid == 0) {
            bval = sval[i];
            if (!bval) s_done = 1;
            bx1 = sx1[i]; by1 = sy1[i]; bx2 = sx2[i]; by2 = sy2[i];
            blab = slab[i];
            s_flag = 0;
        }
        __syncthreads();
        if (s_done) break;                       // uniform (sorted descending)

        const float ai = (bx2 - bx1) * (by2 - by1);
        int sup = 0;
        for (int j = tid; j < i; j += nt) {
            if (!skeep[j] || slab[j] != blab) continue;
            float iw = fmaxf(1e-10f, fminf(bx2, sx2[j]) - fmaxf(bx1, sx1[j]));
            float ih = fmaxf(1e-10f, fminf(by2, sy2[j]) - fmaxf(by1, sy1[j]));
            float inter = iw * ih;
            float aj = (sx2[j] - sx1[j]) * (sy2[j] - sy1[j]);
            float iou = inter / (ai + aj - inter);
            if (iou > NMS_T) { sup = 1; break; }
        }
        if (sup) atomicOr(&s_flag, 1);
        __syncthreads();
        if (tid == 0) skeep[i] = s_flag ? 0 : 1;
        __syncthreads();
    }

    __syncthreads();
    for (int i = tid; i < NS; i += nt) skeep_g[i] = skeep[i];
}

__global__ void scatter_keep(const int* __restrict__ sidx, const int* __restrict__ skeep_g,
                             float* __restrict__ dout)
{
    int i = blockIdx.x * blockDim.x + threadIdx.x;
    if (i >= NS) return;
    int oi = sidx[i];
    if (oi < P) dout[6 * P + oi] = skeep_g[i] ? 1.0f : 0.0f;
}

// ---------------------------------------------------------------------------
extern "C" void kernel_launch(void* const* d_in, const int* in_sizes, int n_in,
                              void* d_out, int out_size, void* d_ws, size_t ws_size,
                              hipStream_t stream)
{
    (void)in_sizes; (void)n_in; (void)out_size; (void)ws_size;

    const float* cls_feat = (const float*)d_in[0];
    const float* reg_feat = (const float*)d_in[1];
    const float* obj_w    = (const float*)d_in[2];
    const float* obj_b    = (const float*)d_in[3];
    const float* cls_w    = (const float*)d_in[4];
    const float* cls_b    = (const float*)d_in[5];
    const float* reg_w    = (const float*)d_in[6];
    const float* reg_b    = (const float*)d_in[7];
    float* dout = (float*)d_out;

    // workspace carve-out (256B aligned)
    char* ws = (char*)d_ws;
    size_t off = 0;
    auto carve = [&](size_t bytes) -> void* {
        void* ptr = ws + off;
        off = (off + bytes + 255) & ~(size_t)255;
        return ptr;
    };
    _Float16* Ffrag = (_Float16*)carve((size_t)FRAG_HALFS * sizeof(_Float16));
    float*    out1  = (float*)carve((size_t)P * 32 * sizeof(float));
    float*    out2  = (float*)carve((size_t)P * 16 * sizeof(float));
    float*    key   = (float*)carve((size_t)NS * sizeof(float));
    int*      sidx  = (int*)carve((size_t)NS * sizeof(int));
    int*      labi  = (int*)carve((size_t)P * sizeof(int));
    float*    sx1   = (float*)carve((size_t)NS * sizeof(float));
    float*    sy1   = (float*)carve((size_t)NS * sizeof(float));
    float*    sx2   = (float*)carve((size_t)NS * sizeof(float));
    float*    sy2   = (float*)carve((size_t)NS * sizeof(float));
    int*      slab  = (int*)carve((size_t)NS * sizeof(int));
    int*      sval  = (int*)carve((size_t)NS * sizeof(int));
    int*      skpg  = (int*)carve((size_t)NS * sizeof(int));

    // 1) pre-swizzle weights into WMMA B-fragment order (f16)
    pack_frags<<<(FRAG_HALFS + 255) / 256, 256, 0, stream>>>(obj_w, cls_w, reg_w, Ffrag);

    // 2) WMMA head GEMM: 576 M-tiles, 8 waves/block -> 72 blocks
    head_gemm<<<(P / 16) / 8, 256, 0, stream>>>(cls_feat, reg_feat, Ffrag, out1, out2);

    // 3) decode + score + sort-key emit
    decode_score<<<(P + 255) / 256, 256, 0, stream>>>(out1, out2, obj_b, cls_b, reg_b,
                                                      dout, key, sidx, labi);
    pad_init<<<((NS - P) + 255) / 256, 256, 0, stream>>>(key, sidx);

    // 4) descending bitonic sort of (key, idx)
    for (int k = 2; k <= NS; k <<= 1)
        for (int j = k >> 1; j > 0; j >>= 1)
            bitonic_pass<<<NS / 256, 256, 0, stream>>>(key, sidx, k, j);

    // 5) gather sorted SoA, greedy NMS, scatter keep mask back
    gather_sorted<<<NS / 256, 256, 0, stream>>>(key, sidx, dout, labi,
                                                sx1, sy1, sx2, sy2, slab, sval);
    nms_greedy<<<1, 512, 0, stream>>>(sx1, sy1, sx2, sy2, slab, sval, skpg);
    scatter_keep<<<NS / 256, 256, 0, stream>>>(sidx, skpg, dout);
}
